// ResidualMemory_50646254354708
// MI455X (gfx1250) — matrix operand order repaired and checked by reading, not compiled, not run on hardware
//
#include <hip/hip_runtime.h>
#include <hip/hip_bf16.h>
#include <math.h>
#include <stdint.h>

typedef _Float16 f16;
typedef _Float16 v16h __attribute__((ext_vector_type(16)));
typedef _Float16 v8h  __attribute__((ext_vector_type(8)));
typedef float    v8f  __attribute__((ext_vector_type(8)));

#define BDIM   4
#define FIN    256
#define SPIX   4096        // 64*64
#define NTOT   16384       // BDIM*SPIX
#define CIN    258
#define KP     288         // CIN padded to multiple of 32
#define QKROWS 1024
#define MROWS  2048        // q rows + k rows
#define NMEM   64
#define SQ     16

// ---------------------------------------------------------------------------
// CDNA5 async global->LDS copy (16B per lane), tracked by ASYNCcnt.
// vdst = LDS byte offset (low 32 bits of generic LDS pointer), vaddr = 64-bit
// global address.  See CDNA5 ISA.
// ---------------------------------------------------------------------------
__device__ __forceinline__ void async_copy16(const void* g, void* l) {
    asm volatile("global_load_async_to_lds_b128 %0, %1, off"
                 :: "v"((uint32_t)(uintptr_t)l), "v"((uint64_t)(uintptr_t)g)
                 : "memory");
}
__device__ __forceinline__ void wait_async5() {
    asm volatile("s_wait_asynccnt 0x5" ::: "memory");
}
__device__ __forceinline__ void wait_async0() {
    asm volatile("s_wait_asynccnt 0x0" ::: "memory");
}

// ---------------------------------------------------------------------------
// Packing kernels
// ---------------------------------------------------------------------------
__global__ __launch_bounds__(256)
void pack_w_kernel(const float* __restrict__ wq, const float* __restrict__ wk,
                   f16* __restrict__ W) {
    int idx = blockIdx.x * 256 + threadIdx.x;      // over 2048*288
    if (idx >= MROWS * KP) return;
    int row = idx / KP, c = idx % KP;
    float v = 0.f;
    if (c < CIN) v = (row < QKROWS) ? wq[row * CIN + c] : wk[(row - QKROWS) * CIN + c];
    W[idx] = (f16)v;
}

__global__ __launch_bounds__(256)
void pack_x_kernel(const float* __restrict__ x, const float* __restrict__ pos,
                   f16* __restrict__ XP) {
    int c = blockIdx.y;                            // 0..287
    int r = blockIdx.x * 256 + threadIdx.x;        // 0..16383
    int b = r >> 12, s = r & 4095;
    float v = 0.f;
    if (c < FIN)          v = x[((size_t)b * FIN + c) * SPIX + s];
    else if (c == FIN)    v = pos[s];
    else if (c == FIN+1)  v = pos[SPIX + s];
    XP[(size_t)r * KP + c] = (f16)v;
}

// vwT[o][n] = sum_f v[n][f] * wout[o][f]
__global__ __launch_bounds__(256)
void pack_vw_kernel(const float* __restrict__ v, const float* __restrict__ wout,
                    f16* __restrict__ VWT) {
    int idx = blockIdx.x * 256 + threadIdx.x;      // 0..16383
    int o = idx >> 6, n = idx & 63;
    float acc = 0.f;
    #pragma unroll 8
    for (int f = 0; f < FIN; ++f) acc += v[n * FIN + f] * wout[o * FIN + f];
    VWT[o * NMEM + n] = (f16)acc;
}

// ---------------------------------------------------------------------------
// GEMM 1: qk[2048][16384] = Wqk[2048][288] * XP^T  (XP stored [N][K])
// workgroup tile 64(M) x 256(N); 8 waves as 2(M) x 4(N), each 32x64 ->
// 8 accumulators; double-buffered async global->LDS staging.
// ---------------------------------------------------------------------------
__global__ __launch_bounds__(256)
void gemm_qk_kernel(const f16* __restrict__ A, const f16* __restrict__ Bm,
                    const float* __restrict__ bq, const float* __restrict__ bk,
                    f16* __restrict__ C) {
    __shared__ f16 As[2][64][32];    // 2 x 4 KB
    __shared__ f16 Bs[2][256][32];   // 2 x 16 KB
    const int tid  = threadIdx.x;
    const int wave = tid >> 5;
    const int lane = tid & 31;
    const int mBase = blockIdx.y * 64;
    const int nBase = blockIdx.x * 256;
    const int wm = (wave >> 2) * 32;   // 2 waves along M
    const int wn = (wave & 3)  * 64;   // 4 waves along N

    // staging assignment: A -> thread covers 16B; B -> thread covers one row (64B)
    const int arow = tid >> 2, asg = (tid & 3) * 8;
    const int brow = tid;
    const f16* agp = A  + (size_t)(mBase + arow) * KP + asg;
    const f16* bgp = Bm + (size_t)(nBase + brow) * KP;

    v8f acc[2][4] = {};

    // prologue: issue tile 0
    async_copy16((const void*)agp, &As[0][arow][asg]);
    #pragma unroll
    for (int c = 0; c < 4; ++c)
        async_copy16((const void*)(bgp + c * 8), &Bs[0][brow][c * 8]);

    const int NK = KP / 32;            // 9 k-tiles
    for (int k = 0; k < NK; ++k) {
        const int cur = k & 1;
        if (k + 1 < NK) {
            const int nxt = cur ^ 1;
            const int k0 = (k + 1) * 32;
            async_copy16((const void*)(agp + k0), &As[nxt][arow][asg]);
            #pragma unroll
            for (int c = 0; c < 4; ++c)
                async_copy16((const void*)(bgp + k0 + c * 8), &Bs[nxt][brow][c * 8]);
            wait_async5();             // tile k's 5 copies retired
        } else {
            wait_async0();
        }
        __syncthreads();

        const int fr  = lane & 15;
        const int akb = (lane >> 4) * 8;
        const int bkb = (lane >> 4) * 8;
        v16h af[2], bf[4];
        #pragma unroll
        for (int i = 0; i < 2; ++i) {
            ((v8h*)&af[i])[0] = *(const v8h*)&As[cur][wm + i*16 + fr][akb];
            ((v8h*)&af[i])[1] = *(const v8h*)&As[cur][wm + i*16 + fr][akb + 16];
        }
        #pragma unroll
        for (int j = 0; j < 4; ++j) {
            ((v8h*)&bf[j])[0] = *(const v8h*)&Bs[cur][wn + j*16 + fr][bkb];
            ((v8h*)&bf[j])[1] = *(const v8h*)&Bs[cur][wn + j*16 + fr][bkb + 16];
        }
        #pragma unroll
        for (int i = 0; i < 2; ++i)
            #pragma unroll
            for (int j = 0; j < 4; ++j)
                acc[i][j] = __builtin_amdgcn_wmma_f32_16x16x32_f16(
                    false, af[i], false, bf[j], (short)0, acc[i][j], false, false);
        __syncthreads();
    }

    // Epilogue: C layout -> row = base + vgpr + 8*(lane>=16), col = base + (lane&15)
    #pragma unroll
    for (int i = 0; i < 2; ++i) {
        int rbase = mBase + wm + i*16 + (lane >> 4) * 8;
        #pragma unroll
        for (int j = 0; j < 4; ++j) {
            int col = nBase + wn + j*16 + (lane & 15);
            #pragma unroll
            for (int jj = 0; jj < 8; ++jj) {
                int row = rbase + jj;
                float bias = (row < QKROWS) ? bq[row] : bk[row - QKROWS];
                C[(size_t)row * NTOT + col] = (f16)(acc[i][j][jj] + bias);
            }
        }
    }
}

// ---------------------------------------------------------------------------
// Attention: per pixel r, normalize q/k over sq=16 per slot, mean qn over
// slots, attn[r][n] = dot(qmean, kn[:,n]).  lane <-> pixel => coalesced.
// ---------------------------------------------------------------------------
__global__ __launch_bounds__(256)
void attn_kernel(const f16* __restrict__ qk, f16* __restrict__ attn) {
    __shared__ f16 tile[256 * NMEM];               // 32 KB staging
    const int tid = threadIdx.x;
    const int r   = blockIdx.x * 256 + tid;

    float qmean[SQ];
    #pragma unroll
    for (int i = 0; i < SQ; ++i) qmean[i] = 0.f;

    for (int m = 0; m < NMEM; ++m) {
        float qv[SQ]; float ss = 0.f;
        #pragma unroll
        for (int kq = 0; kq < SQ; ++kq) {
            qv[kq] = (float)qk[(size_t)(m * SQ + kq) * NTOT + r];
            ss += qv[kq] * qv[kq];
        }
        float inv = 1.0f / fmaxf(sqrtf(ss), 1e-12f);
        #pragma unroll
        for (int kq = 0; kq < SQ; ++kq) qmean[kq] += qv[kq] * inv;
    }
    #pragma unroll
    for (int i = 0; i < SQ; ++i) qmean[i] *= (1.0f / (float)NMEM);

    for (int n = 0; n < NMEM; ++n) {
        float kv[SQ]; float ss = 0.f, dot = 0.f;
        #pragma unroll
        for (int kq = 0; kq < SQ; ++kq) {
            kv[kq] = (float)qk[(size_t)(QKROWS + n * SQ + kq) * NTOT + r];
            ss += kv[kq] * kv[kq];
        }
        float inv = 1.0f / fmaxf(sqrtf(ss), 1e-12f);
        #pragma unroll
        for (int kq = 0; kq < SQ; ++kq) dot += qmean[kq] * kv[kq];
        tile[tid * NMEM + n] = (f16)(dot * inv);
    }
    __syncthreads();
    // vectorized row store: 64 halves = 8 x 16B
    const uint4* src = (const uint4*)&tile[tid * NMEM];
    uint4* dst = (uint4*)&attn[(size_t)r * NMEM];
    #pragma unroll
    for (int c2 = 0; c2 < 8; ++c2) dst[c2] = src[c2];
}

// ---------------------------------------------------------------------------
// GEMM 2: out[r][o] = attn[r][:] . vwT[o][:]  (M=256 over o, N=16384 over r,
// K=64), epilogue adds bout + residual x, writes f32 [B,256,H,W].
// Same 64x256 tile / async double-buffer structure, NK=2.
// ---------------------------------------------------------------------------
__global__ __launch_bounds__(256)
void gemm_out_kernel(const f16* __restrict__ A /*vwT [256][64]*/,
                     const f16* __restrict__ Bm /*attn [16384][64]*/,
                     const float* __restrict__ bout, const float* __restrict__ x,
                     float* __restrict__ out) {
    __shared__ f16 As[2][64][32];
    __shared__ f16 Bs[2][256][32];
    const int tid  = threadIdx.x;
    const int wave = tid >> 5;
    const int lane = tid & 31;
    const int mBase = blockIdx.y * 64;    // 4 blocks over 256 output channels
    const int nBase = blockIdx.x * 256;   // 64 blocks over 16384 pixels
    const int wm = (wave >> 2) * 32;
    const int wn = (wave & 3)  * 64;

    const int arow = tid >> 2, asg = (tid & 3) * 8;
    const int brow = tid;
    const f16* agp = A  + (size_t)(mBase + arow) * NMEM + asg;
    const f16* bgp = Bm + (size_t)(nBase + brow) * NMEM;

    v8f acc[2][4] = {};

    async_copy16((const void*)agp, &As[0][arow][asg]);
    #pragma unroll
    for (int c = 0; c < 4; ++c)
        async_copy16((const void*)(bgp + c * 8), &Bs[0][brow][c * 8]);

    const int NK = NMEM / 32;             // 2 k-tiles
    for (int k = 0; k < NK; ++k) {
        const int cur = k & 1;
        if (k + 1 < NK) {
            const int nxt = cur ^ 1;
            const int k0 = (k + 1) * 32;
            async_copy16((const void*)(agp + k0), &As[nxt][arow][asg]);
            #pragma unroll
            for (int c = 0; c < 4; ++c)
                async_copy16((const void*)(bgp + k0 + c * 8), &Bs[nxt][brow][c * 8]);
            wait_async5();
        } else {
            wait_async0();
        }
        __syncthreads();

        const int fr  = lane & 15;
        const int akb = (lane >> 4) * 8;
        const int bkb = (lane >> 4) * 8;
        v16h af[2], bf[4];
        #pragma unroll
        for (int i = 0; i < 2; ++i) {
            ((v8h*)&af[i])[0] = *(const v8h*)&As[cur][wm + i*16 + fr][akb];
            ((v8h*)&af[i])[1] = *(const v8h*)&As[cur][wm + i*16 + fr][akb + 16];
        }
        #pragma unroll
        for (int j = 0; j < 4; ++j) {
            ((v8h*)&bf[j])[0] = *(const v8h*)&Bs[cur][wn + j*16 + fr][bkb];
            ((v8h*)&bf[j])[1] = *(const v8h*)&Bs[cur][wn + j*16 + fr][bkb + 16];
        }
        #pragma unroll
        for (int i = 0; i < 2; ++i)
            #pragma unroll
            for (int j = 0; j < 4; ++j)
                acc[i][j] = __builtin_amdgcn_wmma_f32_16x16x32_f16(
                    false, af[i], false, bf[j], (short)0, acc[i][j], false, false);
        __syncthreads();
    }

    #pragma unroll
    for (int i = 0; i < 2; ++i) {
        int rbase = mBase + wm + i*16 + (lane >> 4) * 8;
        #pragma unroll
        for (int j = 0; j < 4; ++j) {
            int col = nBase + wn + j*16 + (lane & 15);   // pixel index r
            size_t base = (size_t)(col >> 12) * ((size_t)FIN * SPIX) + (size_t)(col & 4095);
            #pragma unroll
            for (int jj = 0; jj < 8; ++jj) {
                int row = rbase + jj;                    // output channel o
                size_t oidx = base + (size_t)row * SPIX;
                out[oidx] = acc[i][j][jj] + bout[row] + x[oidx];
            }
        }
    }
}

// ---------------------------------------------------------------------------
extern "C" void kernel_launch(void* const* d_in, const int* in_sizes, int n_in,
                              void* d_out, int out_size, void* d_ws, size_t ws_size,
                              hipStream_t stream) {
    const float* x    = (const float*)d_in[0];
    const float* pos  = (const float*)d_in[1];
    const float* wq   = (const float*)d_in[2];
    const float* bq   = (const float*)d_in[3];
    const float* wk   = (const float*)d_in[4];
    const float* bk   = (const float*)d_in[5];
    const float* v    = (const float*)d_in[6];
    const float* wout = (const float*)d_in[7];
    const float* bout = (const float*)d_in[8];
    float* out = (float*)d_out;

    char* ws = (char*)d_ws;
    const size_t off_w    = 0;                                   // 2048*288*2  = 1,179,648
    const size_t off_xp   = off_w    + (size_t)MROWS * KP * 2;   // 16384*288*2 = 9,437,184
    const size_t off_vw   = off_xp   + (size_t)NTOT  * KP * 2;   // 256*64*2    = 32,768
    const size_t off_qk   = off_vw   + (size_t)FIN * NMEM * 2;   // 2048*16384*2= 67,108,864
    const size_t off_attn = off_qk   + (size_t)MROWS * NTOT * 2; // 16384*64*2  = 2,097,152
    f16* Wqk  = (f16*)(ws + off_w);
    f16* XP   = (f16*)(ws + off_xp);
    f16* VWT  = (f16*)(ws + off_vw);
    f16* QK   = (f16*)(ws + off_qk);
    f16* ATTN = (f16*)(ws + off_attn);

    pack_w_kernel <<<dim3((MROWS * KP + 255) / 256), 256, 0, stream>>>(wq, wk, Wqk);
    pack_x_kernel <<<dim3(NTOT / 256, KP),          256, 0, stream>>>(x, pos, XP);
    pack_vw_kernel<<<dim3((FIN * NMEM) / 256),       256, 0, stream>>>(v, wout, VWT);

    gemm_qk_kernel <<<dim3(NTOT / 256, MROWS / 64), 256, 0, stream>>>(Wqk, XP, bq, bk, QK);
    attn_kernel    <<<dim3(NTOT / 256),             256, 0, stream>>>(QK, ATTN);
    gemm_out_kernel<<<dim3(NTOT / 256, FIN / 64),   256, 0, stream>>>(VWT, ATTN, bout, x, out);
}
